// OptimusPrime_66374424592447
// MI455X (gfx1250) — matrix operand order repaired and checked by reading, not compiled, not run on hardware
//
#include <hip/hip_runtime.h>
#include <math.h>
#include <stdint.h>

#define DI static __device__ __forceinline__

typedef __attribute__((ext_vector_type(16))) __bf16 v16bf;
typedef __attribute__((ext_vector_type(8)))  __bf16 v8bf;
typedef __attribute__((ext_vector_type(8)))  float  v8f;

#define D_MODEL 512
#define G3      1536
#define NBOTH   3072
#define B_SZ    8
#define TS      8192
#define T_DS    1024
#define KW      9
#define STRIDE  8
#define MAX_IT  2
#define HALT_EPS 0.01f
#define TIME_PEN 0.01f
#define LN_EPS   1e-6f
#define ROWS    (B_SZ * T_DS)              /* 8192 */
#define WH_PACK_ELEMS (512 * 1536)         /* per-GRU packed wh */

DI float sigf(float x)      { return 1.0f / (1.0f + expf(-x)); }
DI float sig_fast(float x)  { return 1.0f / (1.0f + __expf(-x)); }
DI float tanh_fast(float x) {
  x = fminf(15.f, fmaxf(-15.f, x));
  float e = __expf(2.f * x);
  return (e - 1.f) / (e + 1.f);
}

// ---- WMMA fragment builders (bf16 16x16x32, wave32 layouts per ISA 7.12.2) ----
// A (16x32): lane<16 -> row M=lane, K={0..7,16..23}; lane>=16 -> row M=lane-16, K={8..15,24..31}
DI v16bf load_a_frag(const __bf16* A, int lda, int m0, int k0, int lane) {
  int row = m0 + (lane & 15);
  const __bf16* p = A + (size_t)row * lda + k0 + ((lane >> 4) << 3);
  v16bf a;
#pragma unroll
  for (int i = 0; i < 8; ++i) a[i] = p[i];
#pragma unroll
  for (int i = 0; i < 8; ++i) a[8 + i] = p[16 + i];
  return a;
}

// async global->LDS (CDNA5), 16B per lane; tracked by ASYNCcnt
DI void async_ld128(uint32_t lds_byte_off, const void* gptr) {
  asm volatile("global_load_async_to_lds_b128 %0, %1, off"
               :: "v"(lds_byte_off), "v"(gptr) : "memory");
}
DI void wait_async_le6() { asm volatile("s_wait_asynccnt 0x6" ::: "memory"); }

// ---------------- prep kernels ----------------
// pack a (K x N) f32 matrix (optionally two matrices side-by-side along N at 'split')
// into WMMA B-fragment order: [kc][nt][lane][16 halves]
__global__ void k_pack_b(const float* s0, const float* s1, int split,
                         __bf16* dst, int K, int N) {
  int idx = blockIdx.x * blockDim.x + threadIdx.x;
  if (idx >= K * N) return;
  int i    = idx & 15;
  int lane = (idx >> 4) & 31;
  int rest = idx >> 9;
  int ntiles = N >> 4;
  int nt = rest % ntiles;
  int kc = rest / ntiles;
  int col = nt * 16 + (lane & 15);
  int k   = kc * 32 + ((lane >> 4) << 3) + (i & 7) + ((i >> 3) << 4);
  float v = (col < split) ? s0[(size_t)k * split + col]
                          : s1[(size_t)k * (N - split) + (col - split)];
  dst[idx] = (__bf16)v;
}

__global__ void k_f2bf(const float* src, __bf16* dst, int n) {
  int idx = blockIdx.x * blockDim.x + threadIdx.x;
  if (idx < n) dst[idx] = (__bf16)src[idx];
}

__global__ void k_conv(const float* in, const float* cw, const float* cb, float* state) {
  int idx = blockIdx.x * blockDim.x + threadIdx.x;
  if (idx >= ROWS * D_MODEL) return;
  int c = idx & 511;
  int t = (idx >> 9) & 1023;
  int b = idx >> 19;
  float acc = cb[c];
#pragma unroll
  for (int k = 0; k < KW; ++k) {          // SAME pad: pad_lo=0, pad_hi=1
    int ii = t * STRIDE + k;
    if (ii < TS) acc += in[(size_t)b * TS + ii] * cw[k * D_MODEL + c];
  }
  state[idx] = fmaxf(acc, 0.0f);
}

__global__ void k_init(float* hp, float* rem, float* nup, const int* input_len, int* ds_len) {
  int idx = blockIdx.x * blockDim.x + threadIdx.x;
  if (idx < ROWS) { hp[idx] = 0.f; rem[idx] = 0.f; nup[idx] = 0.f; }
  if (idx < B_SZ) ds_len[idx] = input_len[idx] / STRIDE;
}

// ---------------- WMMA GEMM (xg): C = A @ Bpacked + bi + (bh for z/r gate cols) ----------------
__global__ __launch_bounds__(256) void k_gemm_xg(
    const __bf16* A, const __bf16* Bp, float* C,
    const float* bi0, const float* bi1, const float* bh0, const float* bh1,
    int split, int M, int N, int K) {
  int gwave = (blockIdx.x * blockDim.x + threadIdx.x) >> 5;
  int lane  = threadIdx.x & 31;
  int ntiles = N >> 4;
  int ngrp   = ntiles >> 2;
  int m0 = (gwave / ngrp) << 4;
  int ng = (gwave % ngrp) << 2;          // base n-tile
  if (m0 >= M) return;
  v8f acc[4];
#pragma unroll
  for (int j = 0; j < 4; ++j) acc[j] = (v8f){};
  int nchunks = K >> 5;
  for (int kc = 0; kc < nchunks; ++kc) {
    v16bf a = load_a_frag(A, K, m0, kc * 32, lane);
    const __bf16* bbase = Bp + ((size_t)(kc * ntiles + ng) * 32 + lane) * 16;
    __builtin_prefetch(bbase + (size_t)ntiles * 32 * 16, 0, 0);
#pragma unroll
    for (int j = 0; j < 4; ++j) {
      v16bf b = *(const v16bf*)(bbase + (size_t)j * 32 * 16);
      acc[j] = __builtin_amdgcn_wmma_f32_16x16x32_bf16(false, a, false, b, (short)0, acc[j], false, false);
    }
  }
  int mbase = m0 + ((lane >> 4) << 3);
#pragma unroll
  for (int j = 0; j < 4; ++j) {
    int col = (ng + j) * 16 + (lane & 15);
    int cg = (col < split) ? col : col - split;
    const float* bi = (col < split) ? bi0 : bi1;
    const float* bh = (col < split) ? bh0 : bh1;
    // fold recurrent bias bh into z/r gate pre-activations (cols < 1024 of each GRU);
    // the h-gate bh stays in the scan (it sits inside r*rh)
    float bias = bi[cg] + ((cg < 1024) ? bh[cg] : 0.f);
#pragma unroll
    for (int r = 0; r < 8; ++r) C[(size_t)(mbase + r) * N + col] = acc[j][r] + bias;
  }
}

// merge: inner(8192x512) = [o1|o2](8192x1024)f32 @ packed merge_w + merge_b; 4 N-tiles/wave
__global__ __launch_bounds__(256) void k_gemm_merge(
    const float* o1, const float* o2, const __bf16* Bp, const float* mb, float* C) {
  int gwave = (blockIdx.x * blockDim.x + threadIdx.x) >> 5;
  int lane  = threadIdx.x & 31;
  const int ntiles = D_MODEL >> 4;       // 32
  const int ngrp   = ntiles >> 2;        // 8
  int m0 = (gwave / ngrp) << 4;
  int ng = (gwave % ngrp) << 2;
  if (m0 >= ROWS) return;
  v8f acc[4];
#pragma unroll
  for (int j = 0; j < 4; ++j) acc[j] = (v8f){};
  int arow = m0 + (lane & 15);
  int hi8  = (lane >> 4) << 3;
  for (int kc = 0; kc < 32; ++kc) {
    int kb = kc * 32 + hi8;
    v16bf a;
#pragma unroll
    for (int i = 0; i < 8; ++i) {
      int k = kb + i;
      float v = (k < 512) ? o1[(size_t)arow * 512 + k] : o2[(size_t)arow * 512 + (k - 512)];
      a[i] = (__bf16)v;
    }
#pragma unroll
    for (int i = 0; i < 8; ++i) {
      int k = kb + 16 + i;
      float v = (k < 512) ? o1[(size_t)arow * 512 + k] : o2[(size_t)arow * 512 + (k - 512)];
      a[8 + i] = (__bf16)v;
    }
    const __bf16* bbase = Bp + ((size_t)(kc * ntiles + ng) * 32 + lane) * 16;
    __builtin_prefetch(bbase + (size_t)ntiles * 32 * 16, 0, 0);
#pragma unroll
    for (int j = 0; j < 4; ++j) {
      v16bf b = *(const v16bf*)(bbase + (size_t)j * 32 * 16);
      acc[j] = __builtin_amdgcn_wmma_f32_16x16x32_bf16(false, a, false, b, (short)0, acc[j], false, false);
    }
  }
  int mbase = m0 + ((lane >> 4) << 3);
#pragma unroll
  for (int j = 0; j < 4; ++j) {
    int col = (ng + j) * 16 + (lane & 15);
    float bias = mb[col];
#pragma unroll
    for (int r = 0; r < 8; ++r) C[(size_t)(mbase + r) * D_MODEL + col] = acc[j][r] + bias;
  }
}

// ---------------- persistent GRU scan (grid=2: dir 0 fwd / dir 1 rev) ----------------
// - next step's xg slab (48KB) is async-staged into LDS, overlapped with the WMMAs
// - h kept twice: f32 (recurrence math) + bf16 in A-fragment-friendly rows (two ds_load_b128
//   per K-chunk; rows 8..15 stay zero => free M padding)
__global__ __launch_bounds__(512) void k_gru_scan(
    const float* xg, const __bf16* whp_all,
    const float* bh1, const float* bh2,
    const int* ds_len, float* o1_, float* o2_) {
  __shared__ alignas(16) float  h_lds[8 * 512];          // 16 KB  (f32 recurrent state)
  __shared__ alignas(16) __bf16 h_bf[16 * 512];          // 16 KB  (bf16 copy, 16 rows)
  __shared__ alignas(16) float  rec_lds[8 * G3];         // 48 KB
  __shared__ alignas(16) float  xgbuf[2][8 * G3];        // 96 KB  (double buffer)
  __shared__ alignas(16) float  bhh[512];                //  2 KB  (h-gate recurrent bias)

  int dir = blockIdx.x;
  const __bf16* whp = whp_all + (size_t)dir * WH_PACK_ELEMS;
  const float*  bh  = dir ? bh2 : bh1;
  float*        o   = dir ? o2_ : o1_;
  const float*  xgb = xg + (dir ? G3 : 0);   // column base within 3072-wide rows

  int tid  = threadIdx.x;
  int lane = tid & 31;
  int wave = tid >> 5;
  int lenv[8];
#pragma unroll
  for (int b = 0; b < 8; ++b) lenv[b] = ds_len[b];
  float prev[8];
#pragma unroll
  for (int b = 0; b < 8; ++b) prev[b] = 0.f;
  for (int i = tid; i < 8 * 512; i += 512) h_lds[i] = 0.f;
  for (int i = tid; i < 16 * 512; i += 512) h_bf[i] = (__bf16)0.f;
  if (tid < 512) bhh[tid] = bh[1024 + tid];

  // pre-issue async stage of scan-step 0 (8 rows x 1536 f32 = 3072 16B chunks; 6 per thread)
  {
    int t_x0 = dir ? (T_DS - 1) : 0;
#pragma unroll
    for (int j = 0; j < 6; ++j) {
      int c = j * 512 + tid;
      int b = c / 384;                   // 384 16B-chunks per row
      int w = c - b * 384;
      const float* g = xgb + ((size_t)(b * T_DS + t_x0)) * NBOTH + w * 4;
      async_ld128((uint32_t)(uintptr_t)(&xgbuf[0][b * G3 + w * 4]), g);
    }
  }
  __syncthreads();

  int row = lane & 15;
  int hi8 = (lane >> 4) << 3;
  const __bf16* hrow_bf = h_bf + row * 512;

  for (int t = 0; t < T_DS; ++t) {
    int t_x = dir ? (T_DS - 1 - t) : t;
    // issue async stage for step t+1 into the other buffer (wraps harmlessly at the end)
    {
      int tn   = (t + 1 < T_DS) ? (t + 1) : 0;
      int t_xn = dir ? (T_DS - 1 - tn) : tn;
      float* buf = xgbuf[(t + 1) & 1];
#pragma unroll
      for (int j = 0; j < 6; ++j) {
        int c = j * 512 + tid;
        int b = c / 384;
        int w = c - b * 384;
        const float* g = xgb + ((size_t)(b * T_DS + t_xn)) * NBOTH + w * 4;
        async_ld128((uint32_t)(uintptr_t)(buf + b * G3 + w * 4), g);
      }
    }
    // rec = h @ wh  (16 waves x 6 N-tiles, K=512)
    v8f acc[6];
#pragma unroll
    for (int j = 0; j < 6; ++j) acc[j] = (v8f){};
#pragma unroll 1
    for (int kc = 0; kc < 16; ++kc) {
      int kb = kc * 32 + hi8;
      v8bf lo = *(const v8bf*)(hrow_bf + kb);
      v8bf hi = *(const v8bf*)(hrow_bf + kb + 16);
      v16bf a = __builtin_shufflevector(lo, hi, 0, 1, 2, 3, 4, 5, 6, 7,
                                                8, 9, 10, 11, 12, 13, 14, 15);
#pragma unroll
      for (int j = 0; j < 6; ++j) {
        int nt = wave * 6 + j;
        v16bf bf = *(const v16bf*)(whp + ((size_t)(kc * 96 + nt) * 32 + lane) * 16);
        acc[j] = __builtin_amdgcn_wmma_f32_16x16x32_bf16(false, a, false, bf, (short)0, acc[j], false, false);
      }
    }
    if (lane < 16) {
#pragma unroll
      for (int j = 0; j < 6; ++j) {
        int n0 = (wave * 6 + j) * 16;
#pragma unroll
        for (int r = 0; r < 8; ++r) rec_lds[r * G3 + n0 + lane] = acc[j][r];
      }
    }
    wait_async_le6();        // current step's staged xg complete (in-order), next step's may fly
    __syncthreads();
    // gate math from LDS: thread owns column n for all 8 batch rows
    const float* xb = xgbuf[t & 1];
    int n = tid;
    float bhn = bhh[n];
#pragma unroll 1
    for (int b = 0; b < 8; ++b) {
      float xz  = xb[b * G3 + n];              // already includes bi_z + bh_z
      float xrr = xb[b * G3 + 512 + n];        // already includes bi_r + bh_r
      float xh  = xb[b * G3 + 1024 + n];       // includes bi_h only
      float rz = rec_lds[b * G3 + n];
      float rr = rec_lds[b * G3 + 512 + n];
      float rh = rec_lds[b * G3 + 1024 + n] + bhn;
      float z  = sig_fast(xz + rz);
      float r  = sig_fast(xrr + rr);
      float hh = tanh_fast(xh + r * rh);
      float hold = h_lds[b * 512 + n];
      float hnew = z * hold + (1.f - z) * hh;
      bool  m = (t_x < lenv[b]);
      float hnext = m ? hnew : hold;
      h_lds[b * 512 + n] = hnext;
      h_bf[b * 512 + n]  = (__bf16)hnext;
      prev[b] = m ? hnext : prev[b];
      o[((size_t)(b * T_DS + t)) * D_MODEL + n] = prev[b];
    }
    __syncthreads();
  }
}

// ---------------- LayerNorm + halting (ACT) ----------------
__global__ __launch_bounds__(256) void k_act(
    const float* inner, float* state, const float* ln_g, const float* ln_b,
    const float* act_w, const float* act_b, float* hp, float* rem, float* nup) {
  __shared__ float red[256];
  __shared__ float s_mu, s_rstd, s_uw;
  int row = blockIdx.x;
  int tid = threadIdx.x;
  const float* xin = inner + (size_t)row * D_MODEL;
  float* st = state + (size_t)row * D_MODEL;
  float s0 = st[tid], s1 = st[tid + 256];
  float x0 = xin[tid] + s0, x1 = xin[tid + 256] + s1;

  red[tid] = x0 + x1; __syncthreads();
  for (int s = 128; s > 0; s >>= 1) { if (tid < s) red[tid] += red[tid + s]; __syncthreads(); }
  if (tid == 0) s_mu = red[0] * (1.0f / 512.0f);
  __syncthreads();
  float mu = s_mu, d0 = x0 - mu, d1 = x1 - mu;

  red[tid] = d0 * d0 + d1 * d1; __syncthreads();
  for (int s = 128; s > 0; s >>= 1) { if (tid < s) red[tid] += red[tid + s]; __syncthreads(); }
  if (tid == 0) s_rstd = rsqrtf(red[0] * (1.0f / 512.0f) + LN_EPS);
  __syncthreads();
  float rstd = s_rstd;
  float y0 = d0 * rstd * ln_g[tid] + ln_b[tid];
  float y1 = d1 * rstd * ln_g[tid + 256] + ln_b[tid + 256];

  red[tid] = y0 * act_w[tid] + y1 * act_w[tid + 256]; __syncthreads();
  for (int s = 128; s > 0; s >>= 1) { if (tid < s) red[tid] += red[tid + s]; __syncthreads(); }
  if (tid == 0) {
    float p = sigf(red[0] + act_b[0]);
    float hv = hp[row], rv = rem[row], nv = nup[row];
    float still0 = (hv < 1.0f) ? 1.0f : 0.0f;
    float cond = hv + p * still0;
    float nh   = (cond >  1.0f - HALT_EPS) ? still0 : 0.0f;
    float stl  = (cond <= 1.0f - HALT_EPS) ? still0 : 0.0f;
    hv += p * stl;
    rv += nh * (1.0f - hv);
    hv += nh * rv;
    nv += stl + nh;
    hp[row] = hv; rem[row] = rv; nup[row] = nv;
    s_uw = p * stl + nh * rv;
  }
  __syncthreads();
  float uw = s_uw;
  st[tid]       = y0 * uw + s0 * (1.0f - uw);
  st[tid + 256] = y1 * uw + s1 * (1.0f - uw);
}

__global__ void k_out(const float* state, const float* out_w, const float* out_b, float* out) {
  int idx = blockIdx.x * blockDim.x + threadIdx.x;
  if (idx >= ROWS * 5) return;
  int row = idx / 5, j = idx % 5;
  const float* s = state + (size_t)row * D_MODEL;
  float acc = 0.f;
  for (int d = 0; d < D_MODEL; ++d) acc += s[d] * out_w[d * 5 + j];
  out[idx] = acc + out_b[j];
}

__global__ __launch_bounds__(256) void k_loss(
    const float* nup, const float* rem, const int* ds_len, float* out) {
  __shared__ float red[256];
  int b = blockIdx.x, tid = threadIdx.x;
  int len = ds_len[b];
  float acc = 0.f;
  for (int t = tid; t < T_DS; t += 256)
    if (t < len) acc += nup[b * T_DS + t] + rem[b * T_DS + t];
  red[tid] = acc; __syncthreads();
  for (int s = 128; s > 0; s >>= 1) { if (tid < s) red[tid] += red[tid + s]; __syncthreads(); }
  if (tid == 0) {
    out[ROWS * 5 + b]     = (float)len;                       // ds_len
    out[ROWS * 5 + 8 + b] = red[0] / (float)len * TIME_PEN;   // act_loss
  }
}

extern "C" void kernel_launch(void* const* d_in, const int* in_sizes, int n_in,
                              void* d_out, int out_size, void* d_ws, size_t ws_size,
                              hipStream_t stream) {
  const float* input_data = (const float*)d_in[0];
  const int*   input_len  = (const int*)d_in[1];
  const float* conv_w  = (const float*)d_in[2];
  const float* conv_b  = (const float*)d_in[3];
  const float* g1wi = (const float*)d_in[4];
  const float* g1wh = (const float*)d_in[5];
  const float* g1bi = (const float*)d_in[6];
  const float* g1bh = (const float*)d_in[7];
  const float* g2wi = (const float*)d_in[8];
  const float* g2wh = (const float*)d_in[9];
  const float* g2bi = (const float*)d_in[10];
  const float* g2bh = (const float*)d_in[11];
  const float* merge_w = (const float*)d_in[12];
  const float* merge_b = (const float*)d_in[13];
  const float* ln_g  = (const float*)d_in[14];
  const float* ln_b  = (const float*)d_in[15];
  const float* act_w = (const float*)d_in[16];
  const float* act_b = (const float*)d_in[17];
  const float* out_w = (const float*)d_in[18];
  const float* out_b = (const float*)d_in[19];
  float* out = (float*)d_out;

  char* base = (char*)d_ws;
  size_t off = 0;
  auto wsa = [&](size_t bytes) -> void* {
    void* p = base + off;
    off += (bytes + 255) & ~(size_t)255;
    return p;
  };
  float*   state    = (float*)  wsa((size_t)ROWS * D_MODEL * 4);
  float*   xg       = (float*)  wsa((size_t)ROWS * NBOTH * 4);
  float*   o1       = (float*)  wsa((size_t)ROWS * D_MODEL * 4);
  float*   o2       = (float*)  wsa((size_t)ROWS * D_MODEL * 4);
  float*   inner    = (float*)  wsa((size_t)ROWS * D_MODEL * 4);
  float*   hp       = (float*)  wsa((size_t)ROWS * 4);
  float*   rem      = (float*)  wsa((size_t)ROWS * 4);
  float*   nup      = (float*)  wsa((size_t)ROWS * 4);
  __bf16*  state_bf = (__bf16*) wsa((size_t)ROWS * D_MODEL * 2);
  __bf16*  wip      = (__bf16*) wsa((size_t)512 * NBOTH * 2);       // packed [wi1|wi2]
  __bf16*  whp      = (__bf16*) wsa((size_t)2 * WH_PACK_ELEMS * 2); // packed wh per dir
  __bf16*  mwp      = (__bf16*) wsa((size_t)1024 * 512 * 2);        // packed merge_w
  int*     ds_len   = (int*)    wsa(64);

  // prep: pack every B matrix into WMMA fragment order (bf16)
  k_pack_b<<<(512 * NBOTH + 255) / 256, 256, 0, stream>>>(g1wi, g2wi, G3, wip, 512, NBOTH);
  k_pack_b<<<(WH_PACK_ELEMS + 255) / 256, 256, 0, stream>>>(g1wh, g1wh, G3, whp, 512, G3);
  k_pack_b<<<(WH_PACK_ELEMS + 255) / 256, 256, 0, stream>>>(g2wh, g2wh, G3, whp + WH_PACK_ELEMS, 512, G3);
  k_pack_b<<<(1024 * 512 + 255) / 256, 256, 0, stream>>>(merge_w, merge_w, 512, mwp, 1024, 512);
  k_conv<<<(ROWS * D_MODEL + 255) / 256, 256, 0, stream>>>(input_data, conv_w, conv_b, state);
  k_init<<<(ROWS + 255) / 256, 256, 0, stream>>>(hp, rem, nup, input_len, ds_len);

  for (int it = 0; it < MAX_IT; ++it) {
    k_f2bf<<<(ROWS * D_MODEL + 255) / 256, 256, 0, stream>>>(state, state_bf, ROWS * D_MODEL);
    // xg = state @ [wi1|wi2] + [bi1|bi2] (+ folded bh for z/r)  (M=8192,N=3072,K=512)
    k_gemm_xg<<<(ROWS / 16) * (NBOTH / 64) / 8, 256, 0, stream>>>(
        state_bf, wip, xg, g1bi, g2bi, g1bh, g2bh, G3, ROWS, NBOTH, D_MODEL);
    k_gru_scan<<<2, 512, 0, stream>>>(xg, whp, g1bh, g2bh, ds_len, o1, o2);
    // inner = [o1|o2] @ merge_w + merge_b (M=8192,N=512,K=1024)
    k_gemm_merge<<<(ROWS / 16) * (D_MODEL / 64) / 8, 256, 0, stream>>>(o1, o2, mwp, merge_b, inner);
    k_act<<<ROWS, 256, 0, stream>>>(inner, state, ln_g, ln_b, act_w, act_b, hp, rem, nup);
  }
  k_out<<<(ROWS * 5 + 255) / 256, 256, 0, stream>>>(state, out_w, out_b, out);
  k_loss<<<B_SZ, 256, 0, stream>>>(nup, rem, ds_len, out);
}
